// NCCLoss_26371099197504
// MI455X (gfx1250) — matrix operand order repaired and checked by reading, not compiled, not run on hardware
//
#include <hip/hip_runtime.h>
#include <hip/hip_bf16.h>

// ---------------------------------------------------------------------------
// Fused Sobel-edge-magnitude NCC loss for MI455X (gfx1250, wave32).
//
// conv(27 taps -> 9 filters) per 16 voxels == ONE v_wmma_f32_16x16x32_f16:
//   A (SRC0) = constant weight matrix, M=filters(9 of 16), K=taps(27 of 32)
//   B (SRC1) = im2col voxel data,      K=taps,             N=voxels(16)
//   C        = broadcast 1e-6  (the reference's `g + C*EPSILON` folded into
//                               the WMMA accumulator)
//   D[M][N]: lane m (h=0) holds voxel m filters 0..7, lane m (h=1) filter 8.
// Per-voxel magnitude: in-lane sum of squares + one ds_swizzle(xor 16),
// then raw v_sqrt_f32 (operands are bounded away from denormals by the
// epsilon terms, so the IEEE fix-up sequence is unnecessary).
// Whole loss reduces to 5 streaming sums -> f64 atomics -> finalize kernel.
// ---------------------------------------------------------------------------

typedef __attribute__((ext_vector_type(16))) _Float16 v16h;
typedef __attribute__((ext_vector_type(8)))  float    v8f;

#define NDIM 192
#define VOL  (NDIM * NDIM * NDIM)
#define TZ 8
#define TY 8
#define TX 32
#define PZ (TZ + 2)
#define PY (TY + 2)
#define PX (TX + 2)
#define NTZ (NDIM / TZ)   // 24
#define NTY (NDIM / TY)   // 24
#define NTX (NDIM / TX)   // 6
#define TILES_PER_BATCH (NTZ * NTY * NTX)
#define EPSF 1e-6f

// 9 Sobel kernels (n1=1,n2=2,n3=2), flattened tap index K=(dz*9+dy*3+dx),
// dz,dy,dx in 0..2, dx along the contiguous axis. Matches _sobel_kernels().
__device__ __constant__ float WT[9][27] = {
  // Sx
  {-1,0,1,-2,0,2,-1,0,1,  -2,0,2,-4,0,4,-2,0,2,  -1,0,1,-2,0,2,-1,0,1},
  // Sy
  {-1,-2,-1,0,0,0,1,2,1,  -2,-4,-2,0,0,0,2,4,2,  -1,-2,-1,0,0,0,1,2,1},
  // Sz
  {-1,-2,-1,-2,-4,-2,-1,-2,-1,  0,0,0,0,0,0,0,0,0,  1,2,1,2,4,2,1,2,1},
  // Sd11
  {0,1,2,-1,0,1,-2,-1,0,  0,2,4,-2,0,2,-4,-2,0,  0,1,2,-1,0,1,-2,-1,0},
  // Sd12
  {-2,-1,0,-1,0,1,0,1,2,  -4,-2,0,-2,0,2,0,2,4,  -2,-1,0,-1,0,1,0,1,2},
  // Sd21
  {0,0,0,-1,-2,-1,-2,-4,-2,  1,2,1,0,0,0,-1,-2,-1,  2,4,2,1,2,1,0,0,0},
  // Sd22
  {-2,-4,-2,-1,-2,-1,0,0,0,  -1,-2,-1,0,0,0,1,2,1,  0,0,0,1,2,1,2,4,2},
  // Sd31
  {0,1,2,0,2,4,0,1,2,  -1,0,1,-2,0,2,-1,0,1,  -2,-1,0,-4,-2,0,-2,-1,0},
  // Sd32
  {-2,-1,0,-4,-2,0,-2,-1,0,  -1,0,1,-2,0,2,-1,0,1,  0,1,2,0,2,4,0,1,2},
};

// ds_swizzle xor pattern within wave32 (group-of-32 mode: and=0x1F).
template <int MASK>
__device__ __forceinline__ float swzx(float v) {
  int i = __builtin_bit_cast(int, v);
  i = __builtin_amdgcn_ds_swizzle(i, (MASK << 10) | 0x1F);
  return __builtin_bit_cast(float, i);
}

// Build the B-matrix (data) fragment for one lane.
// B layout (this file's contiguous K order, matched by A): lane = column N, element j holds
// K = 16*h + j. HH is a compile-time half so all LDS offsets fold to
// immediates (consecutive taps merge into wide ds_load).
template <int HH>
__device__ __forceinline__ v16h gatherB(const float* __restrict__ t,
                                        int zc, int yc, int xc) {
  v16h b;
#pragma unroll
  for (int j = 0; j < 16; ++j) {
    const int K = 16 * HH + j;
    float v = 0.0f;
    if (K < 27) {
      const int d0 = K / 9, d1 = (K % 9) / 3, d2 = K % 3;
      v = t[((zc + d0) * PY + (yc + d1)) * PX + (xc + d2)];
    }
    b[j] = (_Float16)v;
  }
  return b;
}

__global__ __launch_bounds__(256)
void edge_ncc_kernel(const float* __restrict__ y,
                     const float* __restrict__ yp,
                     double* __restrict__ acc) {
  __shared__ float sY[PZ * PY * PX];
  __shared__ float sP[PZ * PY * PX];

  const int tid  = threadIdx.x;
  const int lane = tid & 31;
  const int wave = tid >> 5;
  const int h    = lane >> 4;
  const int m    = lane & 15;

  int bid = blockIdx.x;
  const int batch = bid / TILES_PER_BATCH;
  int r = bid % TILES_PER_BATCH;
  const int tz = r / (NTY * NTX);
  r %= (NTY * NTX);
  const int tyi = r / NTX;
  const int txi = r % NTX;
  const int z0 = tz * TZ, y0 = tyi * TY, x0 = txi * TX;

  const float* Yb = y  + (size_t)batch * VOL;
  const float* Pb = yp + (size_t)batch * VOL;

  // Cooperative load of both reflect-padded tiles into LDS.
  for (int i = tid; i < PZ * PY * PX; i += 256) {
    const int lx = i % PX;
    const int t2 = i / PX;
    const int ly = t2 % PY;
    const int lz = t2 / PY;
    int gz = z0 + lz - 1; gz = (gz < 0) ? -gz : ((gz >= NDIM) ? 2 * NDIM - 2 - gz : gz);
    int gy = y0 + ly - 1; gy = (gy < 0) ? -gy : ((gy >= NDIM) ? 2 * NDIM - 2 - gy : gy);
    int gx = x0 + lx - 1; gx = (gx < 0) ? -gx : ((gx >= NDIM) ? 2 * NDIM - 2 - gx : gx);
    const size_t gi = ((size_t)gz * NDIM + gy) * NDIM + gx;
    sY[i] = Yb[gi];
    sP[i] = Pb[gi];
  }
  __syncthreads();

  // Constant A-matrix (weights) fragment: lane = row M (filter), element j
  // holds K = 16h+j  (same contiguous K order as the B fragment).
  v16h wfrag;
#pragma unroll
  for (int j = 0; j < 16; ++j) {
    const int K = 16 * h + j;
    float w = 0.0f;
    if (m < 9 && K < 27) w = WT[m][K];
    wfrag[j] = (_Float16)w;
  }

  float s1 = 0.f, s2 = 0.f, s11 = 0.f, s22 = 0.f, s12 = 0.f;
  // Reference adds C*EPSILON (=1e-6) to every conv output before squaring;
  // fold that bias into the WMMA accumulator operand.
  const v8f ceps = {EPSF, EPSF, EPSF, EPSF, EPSF, EPSF, EPSF, EPSF};

  const int vz = wave;  // one z-plane of the tile per wave (8 waves, TZ=8)
#pragma unroll 1
  for (int vy = 0; vy < TY; ++vy) {
#pragma unroll
    for (int xg = 0; xg < 2; ++xg) {
      const int vx = xg * 16 + m;

      v16h by, bp;
      if (h == 0) {
        by = gatherB<0>(sY, vz, vy, vx);
        bp = gatherB<0>(sP, vz, vy, vx);
      } else {
        by = gatherB<1>(sY, vz, vy, vx);
        bp = gatherB<1>(sP, vz, vy, vx);
      }

      // D = W x B + 1e-6  (g already includes the epsilon bias)
      const v8f dY = __builtin_amdgcn_wmma_f32_16x16x32_f16(
          false, wfrag, false, by, (short)0, ceps, false, false);
      const v8f dP = __builtin_amdgcn_wmma_f32_16x16x32_f16(
          false, wfrag, false, bp, (short)0, ceps, false, false);

      // Per-voxel Sum over 9 filters of g^2 + 1e-6.
      float sy, sp;
      if (h == 0) {         // filters 0..7 in this lane
        sy = 0.f; sp = 0.f;
#pragma unroll
        for (int rr = 0; rr < 8; ++rr) {
          sy += dY[rr] * dY[rr] + EPSF;
          sp += dP[rr] * dP[rr] + EPSF;
        }
      } else {              // filter 8 only (filters 9..15 have zero weights)
        sy = dY[0] * dY[0] + EPSF;
        sp = dP[0] * dP[0] + EPSF;
      }
      sy += swzx<16>(sy);   // combine the two halves -> full 9-filter sum
      sp += swzx<16>(sp);

      if (h == 0) {         // one lane per voxel accumulates
        const float my = __builtin_amdgcn_sqrtf(sy);  // raw v_sqrt_f32
        const float mp = __builtin_amdgcn_sqrtf(sp);
        s1  += my;       s2  += mp;
        s11 += my * my;  s22 += mp * mp;  s12 += my * mp;
      }
    }
  }

  // Wave-level reduction (h==1 lanes contributed zero).
#define RED_(S) S += swzx<1>(S); S += swzx<2>(S); S += swzx<4>(S); \
                S += swzx<8>(S); S += swzx<16>(S);
  RED_(s1) RED_(s2) RED_(s11) RED_(s22) RED_(s12)
#undef RED_

  if (lane == 0) {
    atomicAdd(&acc[0], (double)s1);
    atomicAdd(&acc[1], (double)s2);
    atomicAdd(&acc[2], (double)s11);
    atomicAdd(&acc[3], (double)s22);
    atomicAdd(&acc[4], (double)s12);
  }
}

__global__ void zero_acc_kernel(double* acc) {
  if (threadIdx.x < 5) acc[threadIdx.x] = 0.0;
}

__global__ void finalize_kernel(const double* __restrict__ acc,
                                float* __restrict__ out) {
  const double S1 = acc[0], S2 = acc[1], S11 = acc[2], S22 = acc[3], S12 = acc[4];
  const double n  = (double)(2 * VOL);
  const double c12 = S12 - S1 * S2 / n;
  const double v1  = S11 - S1 * S1 / n;
  const double v2  = S22 - S2 * S2 / n;
  const double ncc = c12 / sqrt(v1 * v2 + 1e-6);
  out[0] = (float)(5000.0 * (1.0 - ncc));
}

extern "C" void kernel_launch(void* const* d_in, const int* in_sizes, int n_in,
                              void* d_out, int out_size, void* d_ws, size_t ws_size,
                              hipStream_t stream) {
  const float* y  = (const float*)d_in[0];
  const float* yp = (const float*)d_in[1];
  double* acc = (double*)d_ws;          // 5 doubles of scratch
  float*  out = (float*)d_out;

  zero_acc_kernel<<<1, 32, 0, stream>>>(acc);
  edge_ncc_kernel<<<dim3(2 * TILES_PER_BATCH), 256, 0, stream>>>(y, yp, acc);
  finalize_kernel<<<1, 1, 0, stream>>>(acc, out);
}
